// GumbelTopoSortAdj_68367289418256
// MI455X (gfx1250) — compile-verified, matrix operand order checked by reading, not run on hardware
//
#include <hip/hip_runtime.h>
#include <math.h>

#define NN   3072
#define WPR  (NN / 32)     // 96 dwords per bit-row

typedef __attribute__((ext_vector_type(16))) _Float16 v16h;
typedef __attribute__((ext_vector_type(8)))  _Float16 v8h;
typedef __attribute__((ext_vector_type(8)))  float    v8f;

// ---------------- Kernel A1: gumbel_logits = logits + (-log(-log(u+eps)+eps)) ----
__global__ void k_gumbel(const float* __restrict__ logits,
                         const float* __restrict__ u,
                         float* __restrict__ gout) {
    int i = blockIdx.x * blockDim.x + threadIdx.x;
    if (i < NN) {
        float g = -logf(-logf(u[i] + 1e-20f) + 1e-20f);
        gout[i] = logits[i] + g;
    }
}

// ---------------- Kernel A2: integer row sums of the 0/1 adjacency -------------
__global__ void k_rowsum(const float* __restrict__ adj, int* __restrict__ rs) {
    __shared__ int sred[256];
    int row = blockIdx.x;
    int t = threadIdx.x;
    int acc = 0;
    const float* rp = adj + (size_t)row * NN;
    for (int j = t; j < NN; j += 256) acc += (rp[j] != 0.0f) ? 1 : 0;
    sred[t] = acc;
    __syncthreads();
    for (int s = 128; s > 0; s >>= 1) {
        if (t < s) sred[t] += sred[t + s];
        __syncthreads();
    }
    if (t == 0) rs[row] = sred[0];
}

// ---------------- Kernel A3: bit-packed transpose: adjT[c] bit i = adj[i][c] ---
__global__ void k_adjT(const float* __restrict__ adj, unsigned* __restrict__ adjT) {
    int tid = blockIdx.x * blockDim.x + threadIdx.x;   // NN*WPR threads total
    int c = tid % NN;          // column (fast-varying -> coalesced loads)
    int w = tid / NN;          // dword index within bit-row
    int base = 32 * w;
    unsigned bits = 0u;
    #pragma unroll 4
    for (int b = 0; b < 32; ++b) {
        bits |= ((adj[(size_t)(base + b) * NN + c] != 0.0f) ? 1u : 0u) << b;
    }
    adjT[(size_t)c * WPR + w] = bits;
}

// ---------------- Kernel B: sequential topo-sort scan (single workgroup) -------
__global__ void __launch_bounds__(1024)
k_scan(const float* __restrict__ gin, const int* __restrict__ rs0,
       const unsigned* __restrict__ adjT, unsigned* __restrict__ maskbits) {
    __shared__ float         g[NN];
    __shared__ int           rowsum[NN];
    __shared__ unsigned char unsel[NN];
    __shared__ float         wv[32];
    __shared__ int           wi[32];
    __shared__ int           sel_idx;

    const int tid  = threadIdx.x;
    const int lane = tid & 31;
    const int wid  = tid >> 5;
    const float NEGINF = -__builtin_huge_valf();

    #pragma unroll
    for (int k = 0; k < 3; ++k) {
        int e = tid + (k << 10);
        g[e] = gin[e];
        rowsum[e] = rs0[e];
        unsel[e] = 1;
    }
    __syncthreads();

    for (int step = 0; step < NN; ++step) {
        // phase 1: per-thread candidate argmax (lowest-index tie-break)
        float bv = NEGINF;
        int   bi = 0x7fffffff;
        int   zf[3];
        #pragma unroll
        for (int k = 0; k < 3; ++k) {
            int e = tid + (k << 10);
            int z = (rowsum[e] == 0) && unsel[e];
            zf[k] = z;
            float v = z ? g[e] : NEGINF;
            if (v > bv || (v == bv && e < bi)) { bv = v; bi = e; }
        }
        // wave32 shuffle reduce
        #pragma unroll
        for (int off = 16; off > 0; off >>= 1) {
            float ov = __shfl_down(bv, off, 32);
            int   oi = __shfl_down(bi, off, 32);
            if (ov > bv || (ov == bv && oi < bi)) { bv = ov; bi = oi; }
        }
        if (lane == 0) { wv[wid] = bv; wi[wid] = bi; }
        __syncthreads();                                    // (a)
        if (wid == 0) {
            bv = wv[lane]; bi = wi[lane];
            #pragma unroll
            for (int off = 16; off > 0; off >>= 1) {
                float ov = __shfl_down(bv, off, 32);
                int   oi = __shfl_down(bi, off, 32);
                if (ov > bv || (ov == bv && oi < bi)) { bv = ov; bi = oi; }
            }
            if (lane == 0) sel_idx = bi;
        }
        __syncthreads();                                    // (b)
        const int idx = sel_idx;

        // record integrated_mask row `idx` = zerodin bits of this step
        #pragma unroll
        for (int k = 0; k < 3; ++k) {
            unsigned m = __builtin_amdgcn_ballot_w32(zf[k] != 0);
            if (lane == 0) maskbits[(size_t)idx * WPR + wid + (k << 5)] = m;
        }
        // state update: deselect idx, subtract adj column idx from rowsums
        if (tid == 0) unsel[idx] = 0;
        #pragma unroll
        for (int k = 0; k < 3; ++k) {
            int e = tid + (k << 10);                 // e>>5 == wid + 32k ; bit == lane
            unsigned bits = adjT[(size_t)idx * WPR + wid + (k << 5)];
            rowsum[e] -= (int)((bits >> lane) & 1u);
        }
        __syncthreads();                                    // (c)
    }
}

// ---------------- Kernel C: log_prob via WMMA masked-exp row reduction ---------
// sum_j mask[i,j]*exp(l_j - l_i) = (sum_j mask[i,j]*exp(l_j)) * exp(-l_i).
// Per wave: 16-row tile; D += A(16x32 f16 masked exp(l_j)) x B(ones); scale after.
__global__ void __launch_bounds__(256)
k_logprob(const float* __restrict__ logits,
          const unsigned* __restrict__ maskbits,
          float* __restrict__ out) {
    __shared__ __align__(16) _Float16 sexp[NN];   // exp(logits[j]) in f16
    const int tid  = threadIdx.x;
    const int lane = tid & 31;
    const int wid  = tid >> 5;

    for (int i = tid; i < NN; i += 256) sexp[i] = (_Float16)expf(logits[i]);
    __syncthreads();

    const int tile  = blockIdx.x * 8 + wid;   // 16-row tile index
    const int r0    = tile * 16;
    const int m     = lane & 15;              // A-matrix row within tile
    const int kbase = (lane >> 4) << 3;       // 0 for lanes 0-15, 8 for lanes 16-31
    const int row   = r0 + m;

    v8f c = {};
    v16h b;
    #pragma unroll
    for (int e = 0; e < 16; ++e) b[e] = (_Float16)1.0f;

    const unsigned* mrow = maskbits + (size_t)row * WPR;
    const v8h* sev = (const v8h*)sexp;        // 16B-aligned vectors of 8 halves
    const _Float16 HZERO = (_Float16)0.0f;

    for (int jc = 0; jc < WPR; ++jc) {
        unsigned bits = mrow[jc];
        if (jc + 1 < WPR) __builtin_prefetch(mrow + jc + 1, 0, 3);
        const int jb = jc << 5;
        // halves jb+kbase..+7 and jb+kbase+16..+23 (16-bit A-matrix VGPR layout)
        v8h lo = sev[(jb + kbase) >> 3];
        v8h hi = sev[(jb + kbase + 16) >> 3];
        v16h a;
        #pragma unroll
        for (int e = 0; e < 8; ++e)
            a[e] = ((bits >> (kbase + e)) & 1u) ? lo[e] : HZERO;
        #pragma unroll
        for (int e = 0; e < 8; ++e)
            a[8 + e] = ((bits >> (kbase + 16 + e)) & 1u) ? hi[e] : HZERO;
        c = __builtin_amdgcn_wmma_f32_16x16x32_f16(
                /*neg_a=*/false, a, /*neg_b=*/false, b,
                /*c_mod=*/(short)0, c, /*reuse_a=*/false, /*reuse_b=*/false);
    }

    // C/D layout: lane 0 holds M=0..7 (N=0) in c[0..7]; lane 16 holds M=8..15.
    if (lane == 0) {
        #pragma unroll
        for (int v = 0; v < 8; ++v) {
            float s = c[v] * expf(-logits[r0 + v]);
            out[r0 + v] = -logf(s + 1e-10f);
        }
    } else if (lane == 16) {
        #pragma unroll
        for (int v = 0; v < 8; ++v) {
            float s = c[v] * expf(-logits[r0 + 8 + v]);
            out[r0 + 8 + v] = -logf(s + 1e-10f);
        }
    }
}

// ---------------- Launch --------------------------------------------------------
extern "C" void kernel_launch(void* const* d_in, const int* in_sizes, int n_in,
                              void* d_out, int out_size, void* d_ws, size_t ws_size,
                              hipStream_t stream) {
    const float* logits = (const float*)d_in[0];
    const float* adj    = (const float*)d_in[1];
    const float* unif   = (const float*)d_in[2];
    float* out = (float*)d_out;          // [0,NN): log_prob ; [NN,2NN): gumbel_logits

    // workspace layout (dwords): adjT[NN*WPR] | rowsum0[NN] | maskbits[NN*WPR]
    unsigned* adjT     = (unsigned*)d_ws;
    int*      rs0      = (int*)(adjT + (size_t)NN * WPR);
    unsigned* maskbits = (unsigned*)(rs0 + NN);

    k_gumbel <<<NN / 256, 256, 0, stream>>>(logits, unif, out + NN);
    k_rowsum <<<NN, 256, 0, stream>>>(adj, rs0);
    k_adjT   <<<(NN * WPR) / 256, 256, 0, stream>>>(adj, adjT);
    k_scan   <<<1, 1024, 0, stream>>>(out + NN, rs0, adjT, maskbits);
    k_logprob<<<NN / 16 / 8, 256, 0, stream>>>(logits, maskbits, out);
}
